// GConv_30313879175647
// MI455X (gfx1250) — matrix operand order repaired and checked by reading, not compiled, not run on hardware
//
#include <hip/hip_runtime.h>

#define N_NODES_C 100000
#define N_EDGES_C 1600000
#define HID 64
#define EDIM 16
#define NGRAPH 512
#define ZZW 192   // 3 layers * 64

typedef __attribute__((ext_vector_type(16))) _Float16 v16h;
typedef __attribute__((ext_vector_type(8)))  float    v8f;

static __device__ __forceinline__ v8f wmma_f16(v16h a, v16h b, v8f c) {
  // D(16x16,f32) = A(16x32,f16) * B(32x16,f16) + C
  return __builtin_amdgcn_wmma_f32_16x16x32_f16(false, a, false, b, (short)0, c, false, false);
}

// ---------------------------------------------------------------------------
// Edge pass: msg = relu(z[src] + edge_attr @ We + be); aggr[dst] += msg
// One wave handles a tile of 16 edges. WMMA: M=16 edges, N=16 hidden, K=32
// (edge dim 16 zero-padded to 32). ZS = compile-time row stride of zin.
// Phased: WMMA x4 -> 32 gather loads -> single wait -> add/relu/atomics.
// ---------------------------------------------------------------------------
template <int ZS>
__global__ __launch_bounds__(256) void edge_msg_kernel(
    const float* __restrict__ ea,
    const int*   __restrict__ srcIdx,
    const int*   __restrict__ dstIdx,
    const float* __restrict__ zin,
    const float* __restrict__ We, const float* __restrict__ be,
    float* __restrict__ aggr)
{
  const int lane = threadIdx.x & 31;
  const int ln16 = lane & 15;
  const int hi   = lane >> 4;                       // 0: lanes 0-15, 1: lanes 16-31
  const int waveId = (blockIdx.x * blockDim.x + threadIdx.x) >> 5;
  const int nWaves = (gridDim.x * blockDim.x) >> 5;

  // Preload B tiles of We (16x64). B layout (32x16, f16): elems 0-7 hold
  // K = hi*8 + i; elems 8-15 (K16..31) are zero padding.
  v16h Bt[4];
  #pragma unroll
  for (int t = 0; t < 4; ++t) {
    const int col = t * 16 + ln16;
    v16h b;
    #pragma unroll
    for (int i = 0; i < 16; ++i)
      b[i] = (i < 8) ? (_Float16)We[(hi * 8 + i) * HID + col] : (_Float16)0.f;
    Bt[t] = b;
  }
  float beReg[4];
  #pragma unroll
  for (int t = 0; t < 4; ++t) beReg[t] = be[t * 16 + ln16];

  const int nTiles = N_EDGES_C / 16;
  for (int tile = waveId; tile < nTiles; tile += nWaves) {
    const int e0 = tile * 16;
    // A (16x32 f16): lanes 0-15: row=lane, elems 0-7 = K0..7; lanes 16-31:
    // row=lane-16, elems 0-7 = K8..15. Elems 8-15 (K16..31) are zero pad.
    const float* ap = ea + (size_t)(e0 + ln16) * EDIM + hi * 8;
    __builtin_prefetch(ap + (size_t)nWaves * 16 * EDIM, 0, 1);
    float4 p0 = *(const float4*)ap;
    float4 p1 = *(const float4*)(ap + 4);
    v16h A = {};
    A[0]=(_Float16)p0.x; A[1]=(_Float16)p0.y; A[2]=(_Float16)p0.z; A[3]=(_Float16)p0.w;
    A[4]=(_Float16)p1.x; A[5]=(_Float16)p1.y; A[6]=(_Float16)p1.z; A[7]=(_Float16)p1.w;

    // C/D layout: lane owns col = t*16 + ln16, rows M = hi*8 + v (v=0..7)
    const int rb = e0 + hi * 8;
    int4 s0 = *(const int4*)(srcIdx + rb);
    int4 s1 = *(const int4*)(srcIdx + rb + 4);
    int4 d0 = *(const int4*)(dstIdx + rb);
    int4 d1 = *(const int4*)(dstIdx + rb + 4);
    // 32-bit element offsets (compile-time strides -> shift/add, no mul_u64)
    const unsigned sOff[8] = {(unsigned)s0.x*ZS, (unsigned)s0.y*ZS, (unsigned)s0.z*ZS,
                              (unsigned)s0.w*ZS, (unsigned)s1.x*ZS, (unsigned)s1.y*ZS,
                              (unsigned)s1.z*ZS, (unsigned)s1.w*ZS};
    const unsigned dOff[8] = {(unsigned)d0.x*HID, (unsigned)d0.y*HID, (unsigned)d0.z*HID,
                              (unsigned)d0.w*HID, (unsigned)d1.x*HID, (unsigned)d1.y*HID,
                              (unsigned)d1.z*HID, (unsigned)d1.w*HID};

    // Phase 1: all 4 WMMAs
    v8f accs[4];
    #pragma unroll
    for (int t = 0; t < 4; ++t) {
      const float bv = beReg[t];
      v8f acc = {bv,bv,bv,bv,bv,bv,bv,bv};
      accs[t] = wmma_f16(A, Bt[t], acc);
    }
    // Phase 2: issue all 32 gathers (pipelined, single wait at first use)
    float zv[32];
    #pragma unroll
    for (int t = 0; t < 4; ++t) {
      const unsigned col = t * 16 + ln16;
      #pragma unroll
      for (int v = 0; v < 8; ++v)
        zv[t * 8 + v] = zin[sOff[v] + col];
    }
    // Phase 3: add + relu + coalesced atomic scatter
    #pragma unroll
    for (int t = 0; t < 4; ++t) {
      const unsigned col = t * 16 + ln16;
      #pragma unroll
      for (int v = 0; v < 8; ++v) {
        float m = accs[t][v] + zv[t * 8 + v];
        m = m > 0.f ? m : 0.f;
        atomicAdd(aggr + dOff[v] + col, m);   // 16 lanes hit 16 consec cols
      }
    }
  }
}

// ---------------------------------------------------------------------------
// Node pass: u = relu( relu((z + aggr) @ W1 + b1) @ W2 + b2 ), in-place in
// aggr; also accumulates per-channel sum / sum-of-squares for BatchNorm.
// One wave per 16-node tile; W1/W2 register-resident as B tiles; LDS
// transpose (C-layout -> A-layout) between the two GEMMs.
// ---------------------------------------------------------------------------
template <int ZS>
__global__ __launch_bounds__(256) void node_mlp_kernel(
    const float* __restrict__ zin,
    float* __restrict__ uo,                  // in: aggr, out: u (in-place)
    const float* __restrict__ W1, const float* __restrict__ b1,
    const float* __restrict__ W2, const float* __restrict__ b2,
    float* __restrict__ stats)               // [0:64)=sum, [64:128)=sumsq
{
  __shared__ float xpose[8][16 * 68];        // per-wave transpose scratch
  __shared__ float bsum[HID];
  __shared__ float bsq[HID];
  const int lane = threadIdx.x & 31;
  const int wib  = threadIdx.x >> 5;
  const int ln16 = lane & 15;
  const int hi   = lane >> 4;
  const int waveId = (blockIdx.x * blockDim.x + threadIdx.x) >> 5;
  const int nWaves = (gridDim.x * blockDim.x) >> 5;

  if (threadIdx.x < HID) { bsum[threadIdx.x] = 0.f; bsq[threadIdx.x] = 0.f; }
  __syncthreads();

  // Preload W1/W2 as B tiles: B(32x16) block kb covers K = kb*32..kb*32+31.
  // elem i<8: K = kb*32 + hi*8 + i ; elem i>=8: K = kb*32 + 16 + hi*8 + (i-8).
  v16h B1r[4][2], B2r[4][2];
  #pragma unroll
  for (int t = 0; t < 4; ++t) {
    const int col = t * 16 + ln16;
    #pragma unroll
    for (int kb = 0; kb < 2; ++kb) {
      v16h w1v, w2v;
      #pragma unroll
      for (int i = 0; i < 16; ++i) {
        const int k = kb * 32 + (i < 8 ? hi * 8 + i : 16 + hi * 8 + (i - 8));
        w1v[i] = (_Float16)W1[k * HID + col];
        w2v[i] = (_Float16)W2[k * HID + col];
      }
      B1r[t][kb] = w1v; B2r[t][kb] = w2v;
    }
  }
  float b1c[4], b2c[4];
  #pragma unroll
  for (int t = 0; t < 4; ++t) { b1c[t] = b1[t*16 + ln16]; b2c[t] = b2[t*16 + ln16]; }

  float psum[4] = {0,0,0,0}, psq[4] = {0,0,0,0};
  float* sw = &xpose[wib][0];

  const int nTiles = N_NODES_C / 16;
  for (int tile = waveId; tile < nTiles; tile += nWaves) {
    const int r0 = tile * 16;

    // hpre = z + aggr, loaded straight into the f16 A layout (K=64 -> A0,A1):
    // lane row = ln16; chunk bases hi*8, 16+hi*8, 32+hi*8, 48+hi*8.
    _Float16 tmp[32];
    {
      const unsigned ro64 = (unsigned)(r0 + ln16) * HID;
      const unsigned roz  = (unsigned)(r0 + ln16) * ZS;
      #pragma unroll
      for (int ch = 0; ch < 4; ++ch) {
        const int cb = ch * 16 + hi * 8;
        float4 za = *(const float4*)(zin + roz + cb);
        float4 zb = *(const float4*)(zin + roz + cb + 4);
        float4 aa = *(const float4*)(uo + ro64 + cb);
        float4 ab = *(const float4*)(uo + ro64 + cb + 4);
        tmp[ch*8+0] = (_Float16)(za.x + aa.x);
        tmp[ch*8+1] = (_Float16)(za.y + aa.y);
        tmp[ch*8+2] = (_Float16)(za.z + aa.z);
        tmp[ch*8+3] = (_Float16)(za.w + aa.w);
        tmp[ch*8+4] = (_Float16)(zb.x + ab.x);
        tmp[ch*8+5] = (_Float16)(zb.y + ab.y);
        tmp[ch*8+6] = (_Float16)(zb.z + ab.z);
        tmp[ch*8+7] = (_Float16)(zb.w + ab.w);
      }
    }
    v16h A0, A1;
    #pragma unroll
    for (int j = 0; j < 16; ++j) { A0[j] = tmp[j]; A1[j] = tmp[16 + j]; }

    // GEMM1: t = relu(hpre @ W1 + b1) -> LDS in C layout
    #pragma unroll
    for (int t = 0; t < 4; ++t) {
      const float bv = b1c[t];
      v8f acc = {bv,bv,bv,bv,bv,bv,bv,bv};
      acc = wmma_f16(A0, B1r[t][0], acc);
      acc = wmma_f16(A1, B1r[t][1], acc);
      const int col = t * 16 + ln16;
      #pragma unroll
      for (int v = 0; v < 8; ++v) {
        float q = acc[v]; q = q > 0.f ? q : 0.f;
        sw[(v + 8 * hi) * 68 + col] = q;
      }
    }
    asm volatile("s_wait_dscnt 0" ::: "memory");  // LDS store -> load, same wave

    // Transpose read back in A layout for GEMM2
    v16h C0, C1;
    {
      const float* sr = sw + ln16 * 68;
      #pragma unroll
      for (int j = 0; j < 8; ++j) {
        C0[j]     = (_Float16)sr[hi * 8 + j];
        C0[j + 8] = (_Float16)sr[16 + hi * 8 + j];
        C1[j]     = (_Float16)sr[32 + hi * 8 + j];
        C1[j + 8] = (_Float16)sr[48 + hi * 8 + j];
      }
    }

    // GEMM2: u = relu(t @ W2 + b2); write in-place + BN partial stats
    #pragma unroll
    for (int t = 0; t < 4; ++t) {
      const float bv = b2c[t];
      v8f acc = {bv,bv,bv,bv,bv,bv,bv,bv};
      acc = wmma_f16(C0, B2r[t][0], acc);
      acc = wmma_f16(C1, B2r[t][1], acc);
      const unsigned col = t * 16 + ln16;
      #pragma unroll
      for (int v = 0; v < 8; ++v) {
        float q = acc[v]; q = q > 0.f ? q : 0.f;
        psum[t] += q; psq[t] += q * q;
        uo[(unsigned)(r0 + v + 8 * hi) * HID + col] = q;
      }
    }
  }

  // Block-level BN reduction: registers -> LDS atomics -> 1 global atomic/ch
  #pragma unroll
  for (int t = 0; t < 4; ++t) {
    atomicAdd(&bsum[t * 16 + ln16], psum[t]);
    atomicAdd(&bsq [t * 16 + ln16], psq[t]);
  }
  __syncthreads();
  if (threadIdx.x < HID) {
    atomicAdd(stats + threadIdx.x,       bsum[threadIdx.x]);
    atomicAdd(stats + HID + threadIdx.x, bsq[threadIdx.x]);
  }
}

// ---------------------------------------------------------------------------
__global__ void bn_final_kernel(float* __restrict__ stats,
                                const float* __restrict__ gamma,
                                const float* __restrict__ beta)
{
  const int c = threadIdx.x;
  if (c < HID) {
    const float inv_n = 1.0f / (float)N_NODES_C;
    float mean = stats[c] * inv_n;
    float var  = stats[HID + c] * inv_n - mean * mean;   // biased variance
    float r    = rsqrtf(var + 1e-5f);
    float sc   = gamma[c] * r;
    stats[128 + c] = sc;                  // scale
    stats[192 + c] = beta[c] - mean * sc; // shift
  }
}

// Vectorized BN apply: one thread = 4 channels (float4 in, float4 out)
__global__ __launch_bounds__(256) void bn_apply_kernel(
    const float* __restrict__ u, const float* __restrict__ ss,
    float* __restrict__ zz, int layer)
{
  int idx = blockIdx.x * 256 + threadIdx.x;        // over N * 16
  if (idx >= N_NODES_C * (HID / 4)) return;
  const int node = idx >> 4;
  const int c4   = (idx & 15) * 4;
  float4 uv = *(const float4*)(u + (unsigned)node * HID + c4);
  float4 sc = *(const float4*)(ss + c4);
  float4 sh = *(const float4*)(ss + HID + c4);
  float4 o;
  o.x = sc.x * uv.x + sh.x;
  o.y = sc.y * uv.y + sh.y;
  o.z = sc.z * uv.z + sh.z;
  o.w = sc.w * uv.w + sh.w;
  *(float4*)(zz + (unsigned)node * ZZW + layer * HID + c4) = o;
}

// Vectorized pool: one thread = 4 channels of one node (float4 load, 4 atomics)
__global__ __launch_bounds__(256) void pool_kernel(
    const float* __restrict__ zz, const int* __restrict__ batch,
    float* __restrict__ g)
{
  int idx = blockIdx.x * 256 + threadIdx.x;        // over N * 48
  if (idx >= N_NODES_C * (ZZW / 4)) return;
  const int node = idx / (ZZW / 4);
  const int c4   = (idx - node * (ZZW / 4)) * 4;
  float4 v = *(const float4*)(zz + (unsigned)node * ZZW + c4);
  float* gp = g + (unsigned)batch[node] * ZZW + c4;
  atomicAdd(gp + 0, v.x);
  atomicAdd(gp + 1, v.y);
  atomicAdd(gp + 2, v.z);
  atomicAdd(gp + 3, v.w);
}

// Vectorized zero: b128 stores
__global__ __launch_bounds__(256) void zero4_kernel(float4* __restrict__ p, int n4)
{
  int i = blockIdx.x * 256 + threadIdx.x;
  if (i < n4) p[i] = make_float4(0.f, 0.f, 0.f, 0.f);
}

// ---------------------------------------------------------------------------
extern "C" void kernel_launch(void* const* d_in, const int* in_sizes, int n_in,
                              void* d_out, int out_size, void* d_ws, size_t ws_size,
                              hipStream_t stream) {
  const float* x    = (const float*)d_in[0];
  const float* ea   = (const float*)d_in[1];
  const float* We   = (const float*)d_in[2];
  const float* be   = (const float*)d_in[3];
  const float* W1   = (const float*)d_in[4];
  const float* b1   = (const float*)d_in[5];
  const float* W2   = (const float*)d_in[6];
  const float* b2   = (const float*)d_in[7];
  const float* gam  = (const float*)d_in[8];
  const float* bet  = (const float*)d_in[9];
  const int* eidx   = (const int*)d_in[10];
  const int* batch  = (const int*)d_in[11];
  const int* srcIdx = eidx;
  const int* dstIdx = eidx + N_EDGES_C;

  float* aggr  = (float*)d_ws;                       // [N,64]  (also holds u)
  float* stats = aggr + (size_t)N_NODES_C * HID;     // 256 floats: sum|sumsq|scale|shift

  float* zz = (float*)d_out;                         // [N,192]
  float* g  = zz + (size_t)N_NODES_C * ZZW;          // [512,192]

  const int zeroN4 = (N_NODES_C * HID + 128) / 4;    // aggr + BN sum/sumsq

  for (int l = 0; l < 3; ++l) {
    const float* zin = (l == 0) ? x : (zz + (size_t)(l - 1) * HID);

    zero4_kernel<<<(zeroN4 + 255) / 256, 256, 0, stream>>>((float4*)aggr, zeroN4);

    if (l == 0) {
      edge_msg_kernel<HID><<<512, 256, 0, stream>>>(
          ea, srcIdx, dstIdx, zin, We + (size_t)l * EDIM * HID, be + l * HID, aggr);
      node_mlp_kernel<HID><<<128, 256, 0, stream>>>(
          zin, aggr, W1 + (size_t)l * HID * HID, b1 + l * HID,
          W2 + (size_t)l * HID * HID, b2 + l * HID, stats);
    } else {
      edge_msg_kernel<ZZW><<<512, 256, 0, stream>>>(
          ea, srcIdx, dstIdx, zin, We + (size_t)l * EDIM * HID, be + l * HID, aggr);
      node_mlp_kernel<ZZW><<<128, 256, 0, stream>>>(
          zin, aggr, W1 + (size_t)l * HID * HID, b1 + l * HID,
          W2 + (size_t)l * HID * HID, b2 + l * HID, stats);
    }
    bn_final_kernel<<<1, 64, 0, stream>>>(stats, gam + l * HID, bet + l * HID);
    bn_apply_kernel<<<(N_NODES_C * (HID / 4) + 255) / 256, 256, 0, stream>>>(
        aggr, stats + 128, zz, l);
  }

  zero4_kernel<<<(NGRAPH * ZZW / 4 + 255) / 256, 256, 0, stream>>>(
      (float4*)g, NGRAPH * ZZW / 4);
  pool_kernel<<<(N_NODES_C * (ZZW / 4) + 255) / 256, 256, 0, stream>>>(
      zz, batch, g);
}
